// BestRQModel_77790447665153
// MI455X (gfx1250) — matrix operand order repaired and checked by reading, not compiled, not run on hardware
//
#include <hip/hip_runtime.h>
#include <hip/hip_bf16.h>
#include <float.h>

// ---------------------------------------------------------------------------
// BestRQ fused loss for MI455X (gfx1250, wave32, WMMA).
// Dominant GEMM [12000,512]x[512,8192] runs on v_wmma_f32_16x16x32_bf16 with
// fp32 accumulation and a fully-fused streaming log-softmax / NLL / argmax
// epilogue (no 393MB logits tensor is ever materialized).
// ---------------------------------------------------------------------------

typedef __attribute__((ext_vector_type(16))) __bf16 v16bf;
typedef __attribute__((ext_vector_type(8)))  __bf16 v8bf;
typedef __attribute__((ext_vector_type(8)))  float  v8f;

constexpr int B_   = 16;
constexpr int T_   = 3000;
constexpr int F_   = 80;
constexpr int NSUB = 750;            // T/STRIDE
constexpr int SF   = 320;            // STRIDE*F
constexpr int E_   = 16;
constexpr int C_   = 8192;
constexpr int D_   = 512;
constexpr int NROWS = B_ * NSUB;     // 12000

// workspace layout (bytes)
constexpr size_t WS_ACC  = 0;                          // double[4]: fpen,nll,mask,corr
constexpr size_t WS_MEAN = 64;                         // float[B*SF]
constexpr size_t WS_RSTD = WS_MEAN + (size_t)B_*SF*4;  // float[B*SF]
constexpr size_t WS_TIDS = WS_RSTD + (size_t)B_*SF*4;  // int[12000]
constexpr size_t WS_BT   = ((WS_TIDS + (size_t)NROWS*4 + 127)/128)*128; // bf16[8192][512]

__device__ inline unsigned short f2bf(float f) {       // round-to-nearest-even
  unsigned int u = __float_as_uint(f);
  unsigned int r = u + 0x7FFFu + ((u >> 16) & 1u);
  return (unsigned short)(r >> 16);
}

// --------------------------- small kernels ---------------------------------

__global__ void zero_acc_kernel(double* acc) {
  if (threadIdx.x < 4) acc[threadIdx.x] = 0.0;
}

__global__ __launch_bounds__(256) void fpen_kernel(const float* __restrict__ feats,
                                                   double* __restrict__ acc) {
  __shared__ float red[256];
  float s = 0.f;
  const size_t n = (size_t)B_ * T_ * F_;
  for (size_t i = (size_t)blockIdx.x * 256 + threadIdx.x; i < n;
       i += (size_t)gridDim.x * 256) {
    float x = feats[i];
    s += x * x;
  }
  red[threadIdx.x] = s;
  __syncthreads();
  for (int o = 128; o > 0; o >>= 1) {
    if ((int)threadIdx.x < o) red[threadIdx.x] += red[threadIdx.x + o];
    __syncthreads();
  }
  if (threadIdx.x == 0) atomicAdd(acc + 0, (double)red[0]);
}

// per-(b,feature-col) mean and 1/(std+1e-5) of the masked stacked features
__global__ __launch_bounds__(256) void stack_stats_kernel(
    const float* __restrict__ feats, const int* __restrict__ flen,
    float* __restrict__ mean, float* __restrict__ rstd) {
  int id = blockIdx.x * 256 + threadIdx.x;
  if (id >= B_ * SF) return;
  int b = id / SF, k = id % SF;
  int cnt = min(NSUB, flen[b] / 4);
  int fsub = k / F_, foff = k % F_;
  const float* p = feats + (size_t)b * T_ * F_ + (size_t)fsub * F_ + foff;
  float s = 0.f, sq = 0.f;
  for (int i = 0; i < cnt; ++i) {
    float x = p[(size_t)i * SF];
    s += x; sq += x * x;
  }
  float mn = s / (float)cnt;
  // sum over ALL 750 rows of (stack-mean)^2, rows >= cnt are zeros
  float var = (sq - 2.f * mn * s + (float)NSUB * mn * mn) / (float)cnt;
  var = fmaxf(var, 0.f);
  mean[id] = mn;
  rstd[id] = 1.0f / (sqrtf(var) + 1e-5f);
}

// project normalized stacked row to 16-d, normalize, argmax dot with 8192
// unit-norm codes (== argmin L2 distance)
__global__ __launch_bounds__(256) void quantize_kernel(
    const float* __restrict__ feats, const int* __restrict__ flen,
    const float* __restrict__ proj,   // [320][16]
    const float* __restrict__ emb,    // [8192][16] (G=1)
    const float* __restrict__ mean, const float* __restrict__ rstd,
    int* __restrict__ tids) {
  __shared__ float sproj[SF * E_];     // 20 KB
  __shared__ float semb[1024 * E_];    // 64 KB
  const int t = threadIdx.x;
  for (int i = t; i < SF * E_; i += 256) sproj[i] = proj[i];

  const int gr = blockIdx.x * 256 + t;
  const bool valid = gr < NROWS;
  const int b = valid ? gr / NSUB : 0;
  const int irow = valid ? gr % NSUB : 0;
  const int cnt = min(NSUB, flen[b] / 4);

  float xs[E_];
#pragma unroll
  for (int e = 0; e < E_; ++e) xs[e] = 0.f;
  __syncthreads();  // sproj ready

  if (valid) {
    const float* fb = feats + (size_t)b * T_ * F_;
    const float* mb = mean + b * SF;
    const float* rb = rstd + b * SF;
    for (int j = 0; j < 4; ++j) {
      const float* frow = fb + (size_t)(4 * irow + j) * F_;
      for (int f = 0; f < F_; ++f) {
        int k = j * F_ + f;
        float x = (irow < cnt) ? frow[f] : 0.f;       // stacked & masked
        float v = (x - mb[k]) * rb[k];                 // normalized
        const float* pr = sproj + k * E_;
#pragma unroll
        for (int e = 0; e < E_; ++e) xs[e] += v * pr[e];
      }
    }
    float sum2 = 0.f;
#pragma unroll
    for (int e = 0; e < E_; ++e) sum2 += xs[e] * xs[e];
    float scale = 1.0f / (sqrtf(sum2) + 1e-8f);
#pragma unroll
    for (int e = 0; e < E_; ++e) xs[e] *= scale;
  }

  float best = -FLT_MAX; int besti = 0;
  for (int cc = 0; cc < C_ / 1024; ++cc) {
    __syncthreads();
    for (int j = 0; j < 64; ++j)
      semb[t + 256 * j] = emb[(size_t)cc * 1024 * E_ + t + 256 * j];
    __syncthreads();
    if (valid) {
      for (int c = 0; c < 1024; ++c) {
        const float* ep = semb + c * E_;
        float d = 0.f;
#pragma unroll
        for (int e = 0; e < E_; ++e) d += xs[e] * ep[e];
        if (d > best) { best = d; besti = cc * 1024 + c; }  // first max kept
      }
    }
  }
  if (valid) tids[gr] = besti;
}

// transpose+convert encoder_top_n_out [512][8192] f32 -> Bt [8192][512] bf16
__global__ __launch_bounds__(256) void convB_kernel(const float* __restrict__ src,
                                                    unsigned short* __restrict__ dst) {
  __shared__ float tile[32][33];
  int bc = blockIdx.x & 255, bd = blockIdx.x >> 8;
  int c0 = bc * 32, d0 = bd * 32;
  int tx = threadIdx.x & 31, ty = threadIdx.x >> 5;
#pragma unroll
  for (int j = 0; j < 4; ++j) {
    int d = d0 + ty + j * 8;
    tile[ty + j * 8][tx] = src[(size_t)d * C_ + c0 + tx];
  }
  __syncthreads();
#pragma unroll
  for (int j = 0; j < 4; ++j) {
    int c = c0 + ty + j * 8;
    dst[(size_t)c * D_ + d0 + tx] = f2bf(tile[tx][ty + j * 8]);
  }
}

// ------------------- fused WMMA GEMM + softmax-CE epilogue ------------------
// WG tile: 64 rows x all 8192 cols, K=512.  8 waves, each wave = 16x64.
__global__ __launch_bounds__(256) void gemm_ce_kernel(
    const float* __restrict__ enc,            // [12000][512] f32
    const __bf16* __restrict__ Bt,            // [8192][512] bf16 (col-major B)
    const int* __restrict__ tids,             // [12000]
    const int* __restrict__ flen,             // [16]
    const unsigned char* __restrict__ cmask,  // [16][750] bool
    double* __restrict__ acc) {
  constexpr int APITCH = 520;  // 512 + 8 bf16 -> kills LDS bank conflicts
  __shared__ __align__(16) unsigned short lds_a[64 * APITCH];  // 65 KB
  __shared__ float sm_m[64], sm_s[64], sm_bv[64], sm_tg[64];
  __shared__ int sm_bi[64];

  const int tid = threadIdx.x;
  const int rowblk = blockIdx.x * 64;

  // stage A: 64x512 f32 -> bf16 in LDS (zero-pad rows past 12000)
  for (int idx = tid; idx < 64 * 512; idx += 256) {
    int r = idx >> 9, k = idx & 511;
    int gr = rowblk + r;
    float v = (gr < NROWS) ? enc[(size_t)gr * D_ + k] : 0.0f;
    lds_a[r * APITCH + k] = f2bf(v);
  }
  __syncthreads();

  const int w = tid >> 5, lane = tid & 31;
  const int msub = w >> 1;             // 0..3 : which 16-row subtile
  const int ngrp = (w & 1) * 64;       // 0/64 : column half of the 128 tile
  const int h = lane >> 4;             // lane half
  const int l15 = lane & 15;
  const int rowbase = msub * 16 + 8 * h;  // 8 rows tracked by this lane

  int tid_r[8];
#pragma unroll
  for (int r = 0; r < 8; ++r) {
    int gr = rowblk + rowbase + r;
    tid_r[r] = (gr < NROWS) ? tids[gr] : -1;
  }

  // per-lane streaming stats for 8 rows
  float mx[8], ss[8], bv[8], tg[8];
  int bi[8];
#pragma unroll
  for (int r = 0; r < 8; ++r) {
    mx[r] = -FLT_MAX; ss[r] = 0.f; bv[r] = -FLT_MAX; bi[r] = 0x7fffffff; tg[r] = -FLT_MAX;
  }

  const __bf16* lA = reinterpret_cast<const __bf16*>(lds_a);

  for (int nt = 0; nt < C_ / 128; ++nt) {
    v8f accv[4];
#pragma unroll
    for (int j = 0; j < 4; ++j)
#pragma unroll
      for (int e = 0; e < 8; ++e) accv[j][e] = 0.0f;

    const int ncol0 = nt * 128 + ngrp;
#pragma unroll 4
    for (int kk = 0; kk < 16; ++kk) {
      const int k0 = kk * 32;
      // A fragment 16x32: lane half h holds K k0+16h..+15
      const __bf16* ap = lA + (msub * 16 + l15) * APITCH + k0 + 16 * h;
      v8bf alo = *reinterpret_cast<const v8bf*>(ap);
      v8bf ahi = *reinterpret_cast<const v8bf*>(ap + 8);
      v16bf a = __builtin_shufflevector(alo, ahi, 0, 1, 2, 3, 4, 5, 6, 7, 8, 9,
                                        10, 11, 12, 13, 14, 15);
#pragma unroll
      for (int j = 0; j < 4; ++j) {
        // B fragment 32x16: lane half h holds K k0+16h..+15 of column n
        const int n = ncol0 + j * 16 + l15;
        const __bf16* bp = Bt + (size_t)n * D_ + k0 + 16 * h;
        v16bf bfrag = *reinterpret_cast<const v16bf*>(bp);
        accv[j] = __builtin_amdgcn_wmma_f32_16x16x32_bf16(
            false, a, false, bfrag, (short)0, accv[j], false, false);
      }
    }
    // streaming epilogue (C layout: elem r of lane = row r+8h, col l15)
#pragma unroll
    for (int j = 0; j < 4; ++j) {
      const int gcol = ncol0 + j * 16 + l15;
#pragma unroll
      for (int r = 0; r < 8; ++r) {
        float v = accv[j][r];
        if (v > bv[r]) { bv[r] = v; bi[r] = gcol; }
        if (gcol == tid_r[r]) tg[r] = v;
        if (v > mx[r]) { ss[r] = ss[r] * __expf(mx[r] - v) + 1.0f; mx[r] = v; }
        else           { ss[r] += __expf(v - mx[r]); }
      }
    }
  }

  // reduce the 16 lanes of each half-wave (they hold the same 8 rows)
#pragma unroll
  for (int r = 0; r < 8; ++r) {
#pragma unroll
    for (int msk = 1; msk < 16; msk <<= 1) {
      float om = __shfl_xor(mx[r], msk);
      float os = __shfl_xor(ss[r], msk);
      float nm = fmaxf(mx[r], om);
      ss[r] = ss[r] * __expf(mx[r] - nm) + os * __expf(om - nm);
      mx[r] = nm;
      float obv = __shfl_xor(bv[r], msk);
      int   obi = __shfl_xor(bi[r], msk);
      if (obv > bv[r] || (obv == bv[r] && obi < bi[r])) { bv[r] = obv; bi[r] = obi; }
      tg[r] = fmaxf(tg[r], __shfl_xor(tg[r], msk));
    }
  }

  // merge the two waves sharing each 16-row subtile, then emit
  if ((w & 1) == 1 && l15 == 0) {
#pragma unroll
    for (int r = 0; r < 8; ++r) {
      int m = rowbase + r;
      sm_m[m] = mx[r]; sm_s[m] = ss[r]; sm_bv[m] = bv[r];
      sm_bi[m] = bi[r]; sm_tg[m] = tg[r];
    }
  }
  __syncthreads();
  if ((w & 1) == 0 && l15 == 0) {
#pragma unroll
    for (int r = 0; r < 8; ++r) {
      int m = rowbase + r;
      float om = sm_m[m], os = sm_s[m];
      float nm = fmaxf(mx[r], om);
      float s = ss[r] * __expf(mx[r] - nm) + os * __expf(om - nm);
      float obv = sm_bv[m]; int obi = sm_bi[m];
      float bvv = bv[r];    int bii = bi[r];
      if (obv > bvv || (obv == bvv && obi < bii)) { bvv = obv; bii = obi; }
      float tgt = fmaxf(tg[r], sm_tg[m]);
      int gr = rowblk + m;
      if (gr < NROWS) {
        int b = gr / NSUB, t = gr % NSUB;
        bool mb = (t < flen[b] / 4) && (cmask[b * NSUB + t] != 0);
        if (mb) {
          float nll = nm + __logf(s) - tgt;  // -(log_softmax at target)
          atomicAdd(acc + 1, (double)nll);
          atomicAdd(acc + 2, 1.0);
          atomicAdd(acc + 3, (bii == tid_r[r]) ? 1.0 : 0.0);
        }
      }
    }
  }
}

__global__ void finalize_kernel(const double* __restrict__ acc,
                                float* __restrict__ out) {
  if (threadIdx.x == 0) {
    double fpen = acc[0] / (double)((size_t)B_ * T_ * F_);
    double msum = acc[2];
    out[0] = (float)(acc[1] / msum + 0.01 * fpen);  // loss
    out[1] = (float)(acc[3] / msum);                // codes_acc (G==1)
    out[2] = (float)fpen;                           // features_pen
    out[3] = (float)msum;                           // num_codes
  }
}

// ---------------------------------------------------------------------------

extern "C" void kernel_launch(void* const* d_in, const int* in_sizes, int n_in,
                              void* d_out, int out_size, void* d_ws, size_t ws_size,
                              hipStream_t stream) {
  (void)in_sizes; (void)n_in; (void)out_size; (void)ws_size;
  const float* feats = (const float*)d_in[0];
  const int* flen = (const int*)d_in[1];
  const unsigned char* cmask = (const unsigned char*)d_in[2];  // jnp bool = 1B
  const float* enc = (const float*)d_in[3];
  const float* proj = (const float*)d_in[4];
  const float* emb = (const float*)d_in[5];
  const float* topn = (const float*)d_in[6];

  char* ws = (char*)d_ws;
  double* acc = (double*)(ws + WS_ACC);
  float* mean = (float*)(ws + WS_MEAN);
  float* rstd = (float*)(ws + WS_RSTD);
  int* tids = (int*)(ws + WS_TIDS);
  unsigned short* Bt = (unsigned short*)(ws + WS_BT);

  zero_acc_kernel<<<1, 32, 0, stream>>>(acc);
  fpen_kernel<<<960, 256, 0, stream>>>(feats, acc);
  stack_stats_kernel<<<(B_ * SF + 255) / 256, 256, 0, stream>>>(feats, flen, mean, rstd);
  quantize_kernel<<<(NROWS + 255) / 256, 256, 0, stream>>>(feats, flen, proj, emb,
                                                           mean, rstd, tids);
  convB_kernel<<<256 * 16, 256, 0, stream>>>(topn, Bt);
  gemm_ce_kernel<<<(NROWS + 63) / 64, 256, 0, stream>>>(
      enc, (const __bf16*)Bt, tids, flen, cmask, acc);
  finalize_kernel<<<1, 32, 0, stream>>>(acc, (float*)d_out);
}
